// Decoder_17360257810657
// MI455X (gfx1250) — compile-verified, matrix operand order checked by reading, not run on hardware
//
#include <hip/hip_runtime.h>
#include <hip/hip_bf16.h>

// ---------------- CDNA5 WMMA types ----------------
typedef __attribute__((ext_vector_type(16))) __bf16 v16bf;
typedef __attribute__((ext_vector_type(8)))  float  v8f;

#define BB    16
#define TIN   512
#define TOUT  150
#define NMEL  80
#define ENC   659
#define PREN  256
#define ATTR  1024
#define ATTD  128
#define NF    32
#define KCONV 31
#define CPAD  15

#define NWG   16
#define NTH   512

// K paddings (multiples of 32)
#define KA_X   928    // 256 + 659 -> 928
#define KD_X   1696   // 1024 + 659 -> 1696
#define KH     1024
#define KMEM   672    // 659 -> 672

__device__ __forceinline__ unsigned short bf16bits(float f) {
  unsigned u = __builtin_bit_cast(unsigned, f);
  unsigned r = u + 0x7FFFu + ((u >> 16) & 1u);   // RNE
  return (unsigned short)(r >> 16);
}
__device__ __forceinline__ float sigf(float x) { return 1.f / (1.f + __expf(-x)); }

// 16-bit A-matrix 16x32 layout (ISA 7.12.2): k -> (laneHi, elem-in-lane)
__device__ __forceinline__ int a_lane_hi(int k)    { return (k >> 3) & 1; }
__device__ __forceinline__ int a_elem_index(int k) {
  int hi = (k >> 3) & 1;
  int kk = k - (hi ? 8 : 0);
  int v  = (kk < 8) ? (kk >> 1) : (4 + ((kk - 16) >> 1));
  return v * 2 + (k & 1);
}
// packed A-tile index for a 16-row activation matrix: [ktile][lane(32)][elem(16)]
__device__ __forceinline__ int a_pack_idx(int m, int k) {
  int kt = k >> 5, kin = k & 31;
  return kt * 512 + ((m & 15) + 16 * a_lane_hi(kin)) * 16 + a_elem_index(kin);
}

// ------ tile GEMM: D(16x16,f32) += A(16xK) * B(KxN-tile), both packed ------
// Apk/Bpk layout: [ktile][lane(32)][elem(16 bf16)]  (lane-contiguous 32B)
__device__ __forceinline__ v8f tile_gemm_packed(const unsigned short* Apk,
                                                const unsigned short* Bpk,
                                                int ktiles, v8f acc, int lane) {
  for (int kt = 0; kt < ktiles; ++kt) {
    v16bf av = *reinterpret_cast<const v16bf*>(Apk + ((size_t)kt * 32 + lane) * 16);
    v16bf bv = *reinterpret_cast<const v16bf*>(Bpk + ((size_t)kt * 32 + lane) * 16);
    if (kt + 1 < ktiles)
      __builtin_prefetch(Bpk + ((size_t)(kt + 1) * 32 + lane) * 16, 0, 0);
    acc = __builtin_amdgcn_wmma_f32_16x16x32_bf16(false, av, false, bv,
                                                  (short)0, acc, false, false);
  }
  return acc;
}

// ---------------- weight pack: f32 [Ksrc][Nsrc] -> bf16 B tiles ----------------
// dst layout: [ntile][ktile][lane(32)][elem(16)]
__global__ void pack_B_kernel(const float* __restrict__ src, unsigned short* __restrict__ dst,
                              int Ksrc, int Nsrc, int Kpad) {
  const int ktiles = Kpad >> 5, ntiles = Nsrc >> 4;
  const long total = (long)ntiles * ktiles * 512;
  for (long i = (long)blockIdx.x * blockDim.x + threadIdx.x; i < total;
       i += (long)gridDim.x * blockDim.x) {
    int e = (int)(i & 15);
    int lane = (int)((i >> 4) & 31);
    long r = i >> 9;
    int kt = (int)(r % ktiles);
    int nt = (int)(r / ktiles);
    int hi = lane >> 4, vv = e >> 1, p = e & 1;
    int kin = (vv < 4 ? 2 * vv : 16 + 2 * (vv - 4)) + (hi ? 8 : 0) + p;
    int k = kt * 32 + kin;
    int n = nt * 16 + (lane & 15);
    float val = (k < Ksrc) ? src[(size_t)k * Nsrc + n] : 0.f;
    dst[i] = bf16bits(val);
  }
}

// ---------------- activation pack: f32 [M][Ksrc] -> bf16 A tiles ----------------
// dst layout: [mtile][ktile][lane(32)][elem(16)]
__global__ void pack_A_kernel(const float* __restrict__ src, unsigned short* __restrict__ dst,
                              int Mtiles, int Ksrc, int Kpad) {
  const int ktiles = Kpad >> 5;
  const long total = (long)Mtiles * ktiles * 512;
  for (long i = (long)blockIdx.x * blockDim.x + threadIdx.x; i < total;
       i += (long)gridDim.x * blockDim.x) {
    int e = (int)(i & 15);
    int lane = (int)((i >> 4) & 31);
    long r = i >> 9;
    int kt = (int)(r % ktiles);
    int mt = (int)(r / ktiles);
    int hi = lane >> 4, vv = e >> 1, p = e & 1;
    int kin = (vv < 4 ? 2 * vv : 16 + 2 * (vv - 4)) + (hi ? 8 : 0) + p;
    int k = kt * 32 + kin;
    int m = mt * 16 + (lane & 15);
    float val = (k < Ksrc) ? src[(size_t)m * Ksrc + k] : 0.f;
    dst[i] = bf16bits(val);
  }
}

// ---------------- WMMA GEMM with pre-packed A and B (processed_memory) ----------------
__global__ void gemm_packed_kernel(const unsigned short* __restrict__ Apk,
                                   const unsigned short* __restrict__ Bpk,
                                   float* __restrict__ C,
                                   int Mtiles, int Ntiles, int ktiles, int ldc) {
  const int lane = threadIdx.x & 31;
  const int wave = ((int)(blockIdx.x * blockDim.x + threadIdx.x)) >> 5;
  const int totalWaves = (int)(gridDim.x * blockDim.x) >> 5;
  for (int job = wave; job < Mtiles * Ntiles; job += totalWaves) {
    int mt = job / Ntiles, nt = job % Ntiles;
    v8f acc = {0.f, 0.f, 0.f, 0.f, 0.f, 0.f, 0.f, 0.f};
    for (int kt = 0; kt < ktiles; ++kt) {
      v16bf a = *reinterpret_cast<const v16bf*>(Apk + (((size_t)mt * ktiles + kt) * 32 + lane) * 16);
      v16bf b = *reinterpret_cast<const v16bf*>(Bpk + (((size_t)nt * ktiles + kt) * 32 + lane) * 16);
      acc = __builtin_amdgcn_wmma_f32_16x16x32_bf16(false, a, false, b, (short)0, acc, false, false);
    }
    int hi = lane >> 4, n = nt * 16 + (lane & 15);
#pragma unroll
    for (int r = 0; r < 8; ++r)
      C[(size_t)(mt * 16 + r + 8 * hi) * ldc + n] = acc[r];
  }
}

// ---------------- prenet (small, VALU) ----------------
__global__ void prenet1_kernel(const float* __restrict__ din, const float* __restrict__ w1,
                               float* __restrict__ h1) {
  for (long i = (long)blockIdx.x * blockDim.x + threadIdx.x; i < (long)TOUT * BB * PREN;
       i += (long)gridDim.x * blockDim.x) {
    int j = (int)(i & 255);
    long r = i >> 8;
    int tt = (int)(r / BB), b = (int)(r % BB);
    float s = 0.f;
    if (tt > 0) {
      const float* x = din + ((size_t)b * NMEL) * TOUT + (tt - 1);
      for (int k = 0; k < NMEL; ++k) s += x[(size_t)k * TOUT] * w1[k * PREN + j];
    }
    h1[i] = fmaxf(s, 0.f);
  }
}
__global__ void prenet2_kernel(const float* __restrict__ h1, const float* __restrict__ w2,
                               float* __restrict__ pre) {
  for (long i = (long)blockIdx.x * blockDim.x + threadIdx.x; i < (long)TOUT * BB * PREN;
       i += (long)gridDim.x * blockDim.x) {
    int j = (int)(i & 255);
    long r = i >> 8;
    const float* hrow = h1 + r * PREN;
    float s = 0.f;
    for (int k = 0; k < PREN; ++k) s += hrow[k] * w2[k * PREN + j];
    pre[i] = fmaxf(s, 0.f);
  }
}

// ---------------- persistent decoder scan ----------------
struct ScanArgs {
  const float* memory;
  const int*   memlen;
  const float* att_bih; const float* att_bhh;
  const float* q_w; const float* v_w; const float* loc_conv_w;
  const float* dec_bih; const float* dec_bhh;
  const float* proj_w; const float* proj_b; const float* gate_w; const float* gate_b;
  const float* pre; const float* pm;
  const unsigned short* attWihPk; const unsigned short* attWhhPk;
  const unsigned short* decWihPk; const unsigned short* decWhhPk;
  const unsigned short* locDensePk;
  unsigned short* convAPk;
  unsigned short* att_hbf; unsigned short* dec_hbf;   // packed A-tile layout
  float* z_att; float* z_dec; float* energies; float* pq; float* loc;
  float* att_h; float* att_c; float* dec_h; float* dec_c;
  float* ctx; float* aw; float* awcum;
  unsigned* sync;
  float* out_mel; float* out_gate; float* out_align;
};

__device__ __forceinline__ void grid_barrier(unsigned* cnt, unsigned* gen, unsigned nwg) {
  __syncthreads();
  if (threadIdx.x == 0) {
    __threadfence();
    unsigned g = *(volatile unsigned*)gen;
    if (atomicAdd(cnt, 1u) + 1u == nwg) {
      *(volatile unsigned*)cnt = 0u;
      __threadfence();
      atomicAdd(gen, 1u);
    } else {
      while (*(volatile unsigned*)gen == g) { __builtin_amdgcn_s_sleep(2); }
    }
    __threadfence();
  }
  __syncthreads();
}

__global__ __launch_bounds__(NTH) void decoder_scan_kernel(ScanArgs a) {
  // packed A-tile staging: [ktile(<=53)][lane(32)][elem(16)] bf16  (54.3 KB)
  __shared__ __align__(32) unsigned short sA[(KD_X / 32) * 512];
  __shared__ float sRed[NTH];
  __shared__ float sAw[TIN];

  const int tid  = threadIdx.x;
  const int wg   = blockIdx.x;
  const int gtid = wg * NTH + tid;
  const int lane = tid & 31;
  const int wav  = tid >> 5;
  const int waveGlobal = gtid >> 5;
  unsigned* cnt = a.sync;
  unsigned* gen = a.sync + 1;

  for (int t = 0; t < TOUT; ++t) {
    // ---- Phase 1: stage [x_t | ctx | 0] (bf16, packed A layout) into LDS ----
    for (int i = tid; i < 16 * KA_X; i += NTH) {
      int m = i / KA_X, k = i % KA_X;
      float v;
      if (k < PREN)            v = a.pre[((size_t)t * BB + m) * PREN + k];
      else if (k < PREN + ENC) v = a.ctx[m * KMEM + (k - PREN)];
      else                     v = 0.f;
      sA[a_pack_idx(m, k)] = bf16bits(v);
    }
    __syncthreads();

    // ---- Phase 2: z_att = [x,ctx]@Wih + h@Whh + b  (WMMA) ----
    {
      const int nt = wg * 16 + wav;                     // 256 N-tiles over N=4096
      v8f acc = {0.f, 0.f, 0.f, 0.f, 0.f, 0.f, 0.f, 0.f};
      acc = tile_gemm_packed(sA, a.attWihPk + (size_t)nt * (KA_X / 32) * 512, KA_X / 32, acc, lane);
      acc = tile_gemm_packed(a.att_hbf, a.attWhhPk + (size_t)nt * (KH / 32) * 512, KH / 32, acc, lane);
      int hi = lane >> 4, n = nt * 16 + (lane & 15);
      float bias = a.att_bih[n] + a.att_bhh[n];
#pragma unroll
      for (int r = 0; r < 8; ++r) a.z_att[(r + 8 * hi) * 4096 + n] = acc[r] + bias;
    }
    grid_barrier(cnt, gen, NWG);

    // ---- Phase 3: att LSTM elementwise; location conv -> packed A ----
    for (int idx = gtid; idx < BB * ATTR; idx += NWG * NTH) {
      int m = idx >> 10, n = idx & 1023;
      float zi = a.z_att[m * 4096 + n];
      float zf = a.z_att[m * 4096 + 1024 + n];
      float zg = a.z_att[m * 4096 + 2048 + n];
      float zo = a.z_att[m * 4096 + 3072 + n];
      float c = sigf(zf) * a.att_c[idx] + sigf(zi) * tanhf(zg);
      float h = sigf(zo) * tanhf(c);
      a.att_c[idx] = c;
      a.att_h[idx] = h;
      a.att_hbf[a_pack_idx(m, n)] = bf16bits(h);
    }
    for (int idx = gtid; idx < BB * TIN * NF; idx += NWG * NTH) {
      int f = idx & 31, bt = idx >> 5;
      int b = bt >> 9, tt = bt & 511;
      const float* w0 = a.loc_conv_w + f * (2 * KCONV);
      const float* w1 = w0 + KCONV;
      const float* s0 = a.aw + b * TIN;
      const float* s1 = a.awcum + b * TIN;
      float s = 0.f;
      for (int kk = 0; kk < KCONV; ++kk) {
        int p = tt + kk - CPAD;
        if (p >= 0 && p < TIN) s += w0[kk] * s0[p] + w1[kk] * s1[p];
      }
      int mt = bt >> 4, mr = bt & 15;
      a.convAPk[(size_t)mt * 512 + (mr + 16 * a_lane_hi(f)) * 16 + a_elem_index(f)] = bf16bits(s);
    }
    grid_barrier(cnt, gen, NWG);

    // ---- Phase 4: loc = convA @ loc_dense (WMMA, M=8192,N=128,K=32); pq = h@q_w ----
    for (int job = waveGlobal; job < (BB * TIN / 16) * (ATTD / 16); job += (NWG * NTH) >> 5) {
      int mt = job >> 3, nt = job & 7;
      v16bf av = *reinterpret_cast<const v16bf*>(a.convAPk + (size_t)mt * 512 + lane * 16);
      v16bf bv = *reinterpret_cast<const v16bf*>(a.locDensePk + (size_t)nt * 512 + lane * 16);
      v8f acc = {0.f, 0.f, 0.f, 0.f, 0.f, 0.f, 0.f, 0.f};
      acc = __builtin_amdgcn_wmma_f32_16x16x32_bf16(false, av, false, bv, (short)0, acc, false, false);
      int hi = lane >> 4, n = nt * 16 + (lane & 15);
#pragma unroll
      for (int r = 0; r < 8; ++r)
        a.loc[(size_t)(mt * 16 + r + 8 * hi) * ATTD + n] = acc[r];
    }
    for (int idx = gtid; idx < BB * ATTD; idx += NWG * NTH) {
      int b = idx >> 7, d = idx & 127;
      float s = 0.f;
      for (int j = 0; j < ATTR; ++j) s += a.att_h[b * ATTR + j] * a.q_w[j * ATTD + d];
      a.pq[idx] = s;
    }
    grid_barrier(cnt, gen, NWG);

    // ---- Phase 5: energies = tanh(pq + loc + pm) @ v_w, masked ----
    for (int idx = gtid; idx < BB * TIN; idx += NWG * NTH) {
      int b = idx >> 9, tt = idx & 511;
      const float* pqr  = a.pq + b * ATTD;
      const float* locr = a.loc + (size_t)idx * ATTD;
      const float* pmr  = a.pm + (size_t)idx * ATTD;
      float s = 0.f;
      for (int d = 0; d < ATTD; ++d) s += a.v_w[d] * tanhf(pqr[d] + locr[d] + pmr[d]);
      if (tt >= a.memlen[b]) s = -1e9f;
      a.energies[idx] = s;
    }
    grid_barrier(cnt, gen, NWG);

    // ---- Phase 6: per-batch softmax + context (WG b handles batch row b) ----
    {
      const int b = wg;
      float e = a.energies[b * TIN + tid];
      sRed[tid] = e; __syncthreads();
      for (int s2 = NTH / 2; s2 > 0; s2 >>= 1) {
        if (tid < s2) sRed[tid] = fmaxf(sRed[tid], sRed[tid + s2]);
        __syncthreads();
      }
      float mx = sRed[0]; __syncthreads();
      float p = __expf(e - mx);
      sRed[tid] = p; __syncthreads();
      for (int s2 = NTH / 2; s2 > 0; s2 >>= 1) {
        if (tid < s2) sRed[tid] += sRed[tid + s2];
        __syncthreads();
      }
      float inv = 1.f / sRed[0]; __syncthreads();
      float w = p * inv;
      sAw[tid] = w;
      a.aw[b * TIN + tid] = w;
      a.awcum[b * TIN + tid] += w;
      a.out_align[((size_t)b * TOUT + t) * TIN + tid] = w;
      __syncthreads();
      for (int e2 = tid; e2 < ENC; e2 += NTH) {
        const float* mrow = a.memory + ((size_t)b * TIN) * ENC + e2;
        float acc = 0.f;
        for (int t2 = 0; t2 < TIN; ++t2) acc += sAw[t2] * mrow[(size_t)t2 * ENC];
        a.ctx[b * KMEM + e2] = acc;
      }
    }
    grid_barrier(cnt, gen, NWG);

    // ---- Phase 7: stage [att_h | ctx | 0] (packed); z_dec WMMA ----
    for (int i = tid; i < 16 * KD_X; i += NTH) {
      int m = i / KD_X, k = i % KD_X;
      float v;
      if (k < ATTR)            v = a.att_h[m * ATTR + k];
      else if (k < ATTR + ENC) v = a.ctx[m * KMEM + (k - ATTR)];
      else                     v = 0.f;
      sA[a_pack_idx(m, k)] = bf16bits(v);
    }
    __syncthreads();
    {
      const int nt = wg * 16 + wav;
      v8f acc = {0.f, 0.f, 0.f, 0.f, 0.f, 0.f, 0.f, 0.f};
      acc = tile_gemm_packed(sA, a.decWihPk + (size_t)nt * (KD_X / 32) * 512, KD_X / 32, acc, lane);
      acc = tile_gemm_packed(a.dec_hbf, a.decWhhPk + (size_t)nt * (KH / 32) * 512, KH / 32, acc, lane);
      int hi = lane >> 4, n = nt * 16 + (lane & 15);
      float bias = a.dec_bih[n] + a.dec_bhh[n];
#pragma unroll
      for (int r = 0; r < 8; ++r) a.z_dec[(r + 8 * hi) * 4096 + n] = acc[r] + bias;
    }
    grid_barrier(cnt, gen, NWG);

    // ---- Phase 8: dec LSTM elementwise ----
    for (int idx = gtid; idx < BB * ATTR; idx += NWG * NTH) {
      int m = idx >> 10, n = idx & 1023;
      float zi = a.z_dec[m * 4096 + n];
      float zf = a.z_dec[m * 4096 + 1024 + n];
      float zg = a.z_dec[m * 4096 + 2048 + n];
      float zo = a.z_dec[m * 4096 + 3072 + n];
      float c = sigf(zf) * a.dec_c[idx] + sigf(zi) * tanhf(zg);
      float h = sigf(zo) * tanhf(c);
      a.dec_c[idx] = c;
      a.dec_h[idx] = h;
      a.dec_hbf[a_pack_idx(m, n)] = bf16bits(h);
    }
    grid_barrier(cnt, gen, NWG);

    // ---- Phase 9: mel / gate projections ----
    if (gtid < BB * NMEL) {
      int b = gtid / NMEL, j = gtid % NMEL;
      float s = a.proj_b[j];
      for (int k = 0; k < ATTR; ++k) s += a.dec_h[b * ATTR + k] * a.proj_w[k * NMEL + j];
      for (int e2 = 0; e2 < ENC; ++e2) s += a.ctx[b * KMEM + e2] * a.proj_w[(ATTR + e2) * NMEL + j];
      a.out_mel[((size_t)b * NMEL + j) * TOUT + t] = s;
    } else if (gtid < BB * NMEL + BB) {
      int b = gtid - BB * NMEL;
      float s = a.gate_b[0];
      for (int k = 0; k < ATTR; ++k) s += a.dec_h[b * ATTR + k] * a.gate_w[k];
      for (int e2 = 0; e2 < ENC; ++e2) s += a.ctx[b * KMEM + e2] * a.gate_w[ATTR + e2];
      a.out_gate[b * TOUT + t] = s;
    }
    grid_barrier(cnt, gen, NWG);
  }
}

// ---------------------------- host launch ----------------------------
extern "C" void kernel_launch(void* const* d_in, const int* in_sizes, int n_in,
                              void* d_out, int out_size, void* d_ws, size_t ws_size,
                              hipStream_t stream) {
  const float* memory     = (const float*)d_in[0];
  const float* din        = (const float*)d_in[1];
  const int*   memlen     = (const int*)d_in[2];
  const float* prenet_w1  = (const float*)d_in[3];
  const float* prenet_w2  = (const float*)d_in[4];
  const float* att_wih    = (const float*)d_in[5];
  const float* att_whh    = (const float*)d_in[6];
  const float* att_bih    = (const float*)d_in[7];
  const float* att_bhh    = (const float*)d_in[8];
  const float* q_w        = (const float*)d_in[9];
  const float* mem_w      = (const float*)d_in[10];
  const float* v_w        = (const float*)d_in[11];
  const float* loc_conv_w = (const float*)d_in[12];
  const float* loc_dense_w= (const float*)d_in[13];
  const float* dec_wih    = (const float*)d_in[14];
  const float* dec_whh    = (const float*)d_in[15];
  const float* dec_bih    = (const float*)d_in[16];
  const float* dec_bhh    = (const float*)d_in[17];
  const float* proj_w     = (const float*)d_in[18];
  const float* proj_b     = (const float*)d_in[19];
  const float* gate_w     = (const float*)d_in[20];
  const float* gate_b     = (const float*)d_in[21];

  char* base = (char*)d_ws;
  size_t off = 0;
  auto alloc = [&](size_t bytes) -> void* {
    void* p = base + off;
    off += (bytes + 255) & ~(size_t)255;
    return p;
  };

  unsigned* sync = (unsigned*)alloc(256);

  // contiguous zero-initialized state block
  const size_t statesBytes =
      (size_t)(4 * BB * ATTR + BB * KMEM + 2 * BB * TIN) * sizeof(float) +
      (size_t)(2 * BB * ATTR) * sizeof(unsigned short);
  char* states = (char*)alloc(statesBytes);
  float* att_h = (float*)states;
  float* att_c = att_h + BB * ATTR;
  float* dec_h = att_c + BB * ATTR;
  float* dec_c = dec_h + BB * ATTR;
  float* ctx   = dec_c + BB * ATTR;
  float* aw    = ctx + BB * KMEM;
  float* awcum = aw + BB * TIN;
  unsigned short* att_hbf = (unsigned short*)(awcum + BB * TIN);
  unsigned short* dec_hbf = att_hbf + BB * ATTR;

  float* z_att    = (float*)alloc((size_t)BB * 4096 * 4);
  float* z_dec    = (float*)alloc((size_t)BB * 4096 * 4);
  float* energies = (float*)alloc((size_t)BB * TIN * 4);
  float* pq       = (float*)alloc((size_t)BB * ATTD * 4);
  float* pre      = (float*)alloc((size_t)TOUT * BB * PREN * 4);
  float* h1       = (float*)alloc((size_t)TOUT * BB * PREN * 4);
  float* pm       = (float*)alloc((size_t)BB * TIN * ATTD * 4);
  float* locbuf   = (float*)alloc((size_t)BB * TIN * ATTD * 4);

  unsigned short* attWihPk  = (unsigned short*)alloc((size_t)KA_X * 4096 * 2);
  unsigned short* attWhhPk  = (unsigned short*)alloc((size_t)KH * 4096 * 2);
  unsigned short* decWihPk  = (unsigned short*)alloc((size_t)KD_X * 4096 * 2);
  unsigned short* decWhhPk  = (unsigned short*)alloc((size_t)KH * 4096 * 2);
  unsigned short* memWPk    = (unsigned short*)alloc((size_t)KMEM * ATTD * 2);
  unsigned short* locDensePk= (unsigned short*)alloc((size_t)NF * ATTD * 2);
  unsigned short* memAPk    = (unsigned short*)alloc((size_t)BB * TIN * KMEM * 2);
  unsigned short* convAPk   = (unsigned short*)alloc((size_t)BB * TIN * NF * 2);

  hipMemsetAsync(sync, 0, 256, stream);
  hipMemsetAsync(states, 0, statesBytes, stream);

  // weight / activation packing (bf16 WMMA tile layouts)
  pack_B_kernel<<<512, 256, 0, stream>>>(att_wih, attWihPk, PREN + ENC, 4096, KA_X);
  pack_B_kernel<<<512, 256, 0, stream>>>(att_whh, attWhhPk, ATTR, 4096, KH);
  pack_B_kernel<<<512, 256, 0, stream>>>(dec_wih, decWihPk, ATTR + ENC, 4096, KD_X);
  pack_B_kernel<<<512, 256, 0, stream>>>(dec_whh, decWhhPk, ATTR, 4096, KH);
  pack_B_kernel<<<64, 256, 0, stream>>>(mem_w, memWPk, ENC, ATTD, KMEM);
  pack_B_kernel<<<16, 256, 0, stream>>>(loc_dense_w, locDensePk, NF, ATTD, NF);
  pack_A_kernel<<<512, 256, 0, stream>>>(memory, memAPk, (BB * TIN) / 16, ENC, KMEM);

  // prenet
  prenet1_kernel<<<512, 256, 0, stream>>>(din, prenet_w1, h1);
  prenet2_kernel<<<512, 256, 0, stream>>>(h1, prenet_w2, pre);

  // processed_memory = memory @ mem_w  (WMMA)
  gemm_packed_kernel<<<256, 256, 0, stream>>>(memAPk, memWPk, pm,
                                              (BB * TIN) / 16, ATTD / 16, KMEM / 32, ATTD);

  float* out = (float*)d_out;
  ScanArgs sa;
  sa.memory = memory; sa.memlen = memlen;
  sa.att_bih = att_bih; sa.att_bhh = att_bhh;
  sa.q_w = q_w; sa.v_w = v_w; sa.loc_conv_w = loc_conv_w;
  sa.dec_bih = dec_bih; sa.dec_bhh = dec_bhh;
  sa.proj_w = proj_w; sa.proj_b = proj_b; sa.gate_w = gate_w; sa.gate_b = gate_b;
  sa.pre = pre; sa.pm = pm;
  sa.attWihPk = attWihPk; sa.attWhhPk = attWhhPk;
  sa.decWihPk = decWihPk; sa.decWhhPk = decWhhPk;
  sa.locDensePk = locDensePk;
  sa.convAPk = convAPk;
  sa.att_hbf = att_hbf; sa.dec_hbf = dec_hbf;
  sa.z_att = z_att; sa.z_dec = z_dec; sa.energies = energies; sa.pq = pq; sa.loc = locbuf;
  sa.att_h = att_h; sa.att_c = att_c; sa.dec_h = dec_h; sa.dec_c = dec_c;
  sa.ctx = ctx; sa.aw = aw; sa.awcum = awcum;
  sa.sync = sync;
  sa.out_mel = out;
  sa.out_gate = out + (size_t)BB * NMEL * TOUT;
  sa.out_align = out + (size_t)BB * NMEL * TOUT + (size_t)BB * TOUT;

  decoder_scan_kernel<<<NWG, NTH, 0, stream>>>(sa);
}
